// PTAttention3D_67697274519578
// MI455X (gfx1250) — compile-verified, matrix operand order checked by reading, not run on hardware
//
#include <hip/hip_runtime.h>

typedef __attribute__((ext_vector_type(16))) _Float16 v16h;
typedef __attribute__((ext_vector_type(8)))  _Float16 v8h;
typedef __attribute__((ext_vector_type(2)))  __fp16   pk2h;  // cvt_pkrtz result type
typedef __attribute__((ext_vector_type(8)))  float    v8f;
typedef __attribute__((ext_vector_type(4)))  float    v4f;
typedef __attribute__((ext_vector_type(4)))  int      v4i;

#define B_   2
#define C_   128
#define T_   4
#define NH_  32
#define N_   1024     // T * 16 * 16 patch tokens
#define TBL_ 6727     // (2T-1)*(2h-1)*(2w-1) = 7*31*31
#define SPG_ 65536.0f // elements per GroupNorm group: 4 channels * T*H*W

// ---------------------------------------------------------------------------
// Kernel 0: zero the atomic stats accumulator (harness poisons ws with 0xAA).
// ---------------------------------------------------------------------------
__global__ void k_zero(float* __restrict__ p, int n) {
    int i = blockIdx.x * blockDim.x + threadIdx.x;
    if (i < n) p[i] = 0.0f;
}

// ---------------------------------------------------------------------------
// Kernel 1: fused q/k/v projection, patch pooling (q,k), raw-f16 V pack in
// [b][g][dp=dd*16+pix][m] layout, and GroupNorm sum/sumsq atomics.
// One block per (b, patch). 512 threads: tid = og*16 + pix, og==group==head.
// ---------------------------------------------------------------------------
__global__ __launch_bounds__(512)
void k_proj(const float* __restrict__ x,  const float* __restrict__ Wq,
            const float* __restrict__ Wk, const float* __restrict__ Wv,
            float* __restrict__ qp_raw, float* __restrict__ kp_raw,
            _Float16* __restrict__ vh, float* __restrict__ statsRaw)
{
    __shared__ float sX[16][C_];   // [pixel][channel], 8 KB

    const int blk = blockIdx.x;          // b*1024 + m
    const int b   = blk >> 10;
    const int m   = blk & 1023;
    const int tt  = m >> 8;
    const int yh  = (m >> 4) & 15;
    const int xw  = m & 15;
    const int tid = threadIdx.x;

    { // stage x tile: thread (c, sh) loads 4 contiguous W-pixels
        const int c  = tid >> 2;
        const int sh = tid & 3;
        const float* src = x + ((size_t)((b * C_ + c) * T_ + tt)) * 4096
                             + (yh * 4 + sh) * 64 + xw * 4;
        v4f xv = *(const v4f*)src;
        const int pb = sh * 4;
        sX[pb + 0][c] = xv[0]; sX[pb + 1][c] = xv[1];
        sX[pb + 2][c] = xv[2]; sX[pb + 3][c] = xv[3];
    }
    __syncthreads();

    const int pix = tid & 15;
    const int og  = tid >> 4;            // 0..31 == group == head
    float aq[4] = {0,0,0,0}, ak[4] = {0,0,0,0}, av[4] = {0,0,0,0};

    const float* xrow = &sX[pix][0];
    for (int c = 0; c < C_; c += 4) {
        v4f xv = *(const v4f*)(xrow + c);
#pragma unroll
        for (int oo = 0; oo < 4; ++oo) {
            const int o = og * 4 + oo;
            v4f wq = *(const v4f*)(Wq + (size_t)o * C_ + c);
            v4f wk = *(const v4f*)(Wk + (size_t)o * C_ + c);
            v4f wv = *(const v4f*)(Wv + (size_t)o * C_ + c);
            aq[oo] += wq[0]*xv[0] + wq[1]*xv[1] + wq[2]*xv[2] + wq[3]*xv[3];
            ak[oo] += wk[0]*xv[0] + wk[1]*xv[1] + wk[2]*xv[2] + wk[3]*xv[3];
            av[oo] += wv[0]*xv[0] + wv[1]*xv[1] + wv[2]*xv[2] + wv[3]*xv[3];
        }
    }

    // raw f16 V pack: vh[((b*NH+g)*64 + dd*16 + pix)*N + m]
#pragma unroll
    for (int oo = 0; oo < 4; ++oo) {
        size_t vi = ((size_t)((b * NH_ + og) * 64 + oo * 16 + pix)) * N_ + m;
        vh[vi] = (_Float16)av[oo];
    }

    // per-thread stat partials (sum + sumsq over this thread's 4 channels)
    float sq = 0, q2 = 0, sk = 0, k2 = 0, sv = 0, v2 = 0;
#pragma unroll
    for (int oo = 0; oo < 4; ++oo) {
        sq += aq[oo]; q2 += aq[oo]*aq[oo];
        sk += ak[oo]; k2 += ak[oo]*ak[oo];
        sv += av[oo]; v2 += av[oo]*av[oo];
    }

    // reduce over the 16 pixel lanes (xor<16 stays inside each wave32 half)
#pragma unroll
    for (int off = 1; off < 16; off <<= 1) {
#pragma unroll
        for (int oo = 0; oo < 4; ++oo) {
            aq[oo] += __shfl_xor(aq[oo], off, 32);
            ak[oo] += __shfl_xor(ak[oo], off, 32);
        }
        sq += __shfl_xor(sq, off, 32); q2 += __shfl_xor(q2, off, 32);
        sk += __shfl_xor(sk, off, 32); k2 += __shfl_xor(k2, off, 32);
        sv += __shfl_xor(sv, off, 32); v2 += __shfl_xor(v2, off, 32);
    }

    if (pix == 0) {
        size_t qi = ((size_t)(b * N_ + m)) * C_ + og * 4;
        v4f pq = {aq[0], aq[1], aq[2], aq[3]};   // patch SUM (mean applied later)
        v4f pk = {ak[0], ak[1], ak[2], ak[3]};
        *(v4f*)(qp_raw + qi) = pq;
        *(v4f*)(kp_raw + qi) = pk;
        float* sQ = statsRaw + ((size_t)((0 * B_ + b) * NH_ + og)) * 2;
        float* sK = statsRaw + ((size_t)((1 * B_ + b) * NH_ + og)) * 2;
        float* sV = statsRaw + ((size_t)((2 * B_ + b) * NH_ + og)) * 2;
        unsafeAtomicAdd(sQ + 0, sq); unsafeAtomicAdd(sQ + 1, q2);
        unsafeAtomicAdd(sK + 0, sk); unsafeAtomicAdd(sK + 1, k2);
        unsafeAtomicAdd(sV + 0, sv); unsafeAtomicAdd(sV + 1, v2);
    }
}

// ---------------------------------------------------------------------------
// Kernel 2: finalize GroupNorm stats -> (mu, rstd) per (proj, b, g)
// ---------------------------------------------------------------------------
__global__ void k_stats(const float* __restrict__ raw, float* __restrict__ fin) {
    int i = blockIdx.x * blockDim.x + threadIdx.x;
    if (i < 3 * B_ * NH_) {
        float s  = raw[2 * i + 0];
        float ss = raw[2 * i + 1];
        float mu  = s / SPG_;
        float var = ss / SPG_ - mu * mu;
        fin[2 * i + 0] = mu;
        fin[2 * i + 1] = rsqrtf(var + 1e-5f);
    }
}

// ---------------------------------------------------------------------------
// Kernel 3: normalize pooled q,k (GN affine commutes with patch mean).
// Fold the logit scale 1/sqrt(d)=0.5 into q.
// ---------------------------------------------------------------------------
__global__ __launch_bounds__(256)
void k_norm(const float* __restrict__ qp_raw, const float* __restrict__ kp_raw,
            const float* __restrict__ gq_g, const float* __restrict__ gq_b,
            const float* __restrict__ gk_g, const float* __restrict__ gk_b,
            const float* __restrict__ fin,
            float* __restrict__ qp_n, float* __restrict__ kp_n)
{
    int i = blockIdx.x * blockDim.x + threadIdx.x;   // over B*N*C
    if (i >= B_ * N_ * C_) return;
    const int c = i & (C_ - 1);
    const int b = i >> 17;                           // N*C = 131072
    const int g = c >> 2;
    const float zq = qp_raw[i] * (1.0f / 16.0f);
    const float zk = kp_raw[i] * (1.0f / 16.0f);
    const float muq = fin[((0 * B_ + b) * NH_ + g) * 2 + 0];
    const float rsq = fin[((0 * B_ + b) * NH_ + g) * 2 + 1];
    const float muk = fin[((1 * B_ + b) * NH_ + g) * 2 + 0];
    const float rsk = fin[((1 * B_ + b) * NH_ + g) * 2 + 1];
    qp_n[i] = ((zq - muq) * rsq * gq_g[c] + gq_b[c]) * 0.5f;
    kp_n[i] = (zk - muk) * rsk * gk_g[c] + gk_b[c];
}

// ---------------------------------------------------------------------------
// Kernel 3b: bake the V GroupNorm affine into the f16 V buffer (in place),
// so the attention inner loop has no per-fragment VALU on V.
// vh layout: [b][g][dp][m]; channel for a row = g*4 + (dp>>4).
// ---------------------------------------------------------------------------
__global__ __launch_bounds__(256)
void k_vnorm(_Float16* __restrict__ vh, const float* __restrict__ fin,
             const float* __restrict__ gv_g, const float* __restrict__ gv_b)
{
    const int total = B_ * NH_ * 64 * (N_ / 8);
    int i = blockIdx.x * blockDim.x + threadIdx.x;   // one v8h per thread
    if (i >= total) return;
    const int hbase = i * 8;
    const int row = hbase >> 10;          // (b*NH+g)*64 + dp
    const int dp  = row & 63;
    const int g   = (row >> 6) & 31;
    const int b   = row >> 11;
    const int c   = g * 4 + (dp >> 4);
    const float muv = fin[((2 * B_ + b) * NH_ + g) * 2 + 0];
    const float rsv = fin[((2 * B_ + b) * NH_ + g) * 2 + 1];
    const float a  = rsv * gv_g[c];
    const float bb = gv_b[c] - muv * a;
    v8h v = *(v8h*)(vh + hbase);
#pragma unroll
    for (int e = 0; e < 8; ++e) v[e] = (_Float16)((float)v[e] * a + bb);
    *(v8h*)(vh + hbase) = v;
}

// ---------------------------------------------------------------------------
// Async staging of one 32-key x 64-dp V chunk (4 KB) into LDS, shared by the
// 4 waves of the block.  Per-lane global_load_async_to_lds_b128 (ASYNCcnt),
// 2 segments per thread.  LDS layout: [dp][k_local(32)] halves, 64 B rows.
// ---------------------------------------------------------------------------
__device__ __forceinline__
void stage_v(const _Float16* __restrict__ gsrc, _Float16* ldst, int tid)
{
#pragma unroll
    for (int it = 0; it < 2; ++it) {
        const int idx = tid + it * 128;        // 0..255 segments of 16 B
        const int dp  = idx >> 2;
        const int seg = idx & 3;
        const _Float16* g = gsrc + (size_t)dp * N_ + seg * 8;
        const _Float16* l = ldst + dp * 32 + seg * 8;
        const unsigned lofs = (unsigned)(unsigned long long)l;  // LDS byte addr
        const unsigned long long ga = (unsigned long long)g;
        asm volatile("global_load_async_to_lds_b128 %0, %1, off"
                     :: "v"(lofs), "v"(ga) : "memory");
    }
}

// ---------------------------------------------------------------------------
// Kernel 4: fused flash attention.  Block = 128 threads (4 waves), one block
// per (b, g, 64-query tile); each wave owns 16 queries.  Per 32-key chunk:
// VALU logits (+LDS rel-bias gather), online softmax (xor-16 combine),
// pkrtz-pack P directly into the WMMA A layout, 4x v_wmma_f32_16x16x32_f16
// against V fragments read from the async-staged, double-buffered LDS tile.
// ---------------------------------------------------------------------------
__global__ __launch_bounds__(128)
void k_attn(const float* __restrict__ qp_n, const float* __restrict__ kp_n,
            const _Float16* __restrict__ vh,
            const float* __restrict__ rel_table, const int* __restrict__ rel_index,
            float* __restrict__ out)
{
    __shared__ float sKP[N_ * 4];                      // kp for this (b,g): 16 KB
    __shared__ float sTab[TBL_];                       // rel_table row: ~27 KB
    __shared__ __align__(16) _Float16 sV[2][64 * 32];  // double-buffered V chunk: 2x4 KB

    const int blk  = blockIdx.x;                 // b*512 + g*16 + tile
    const int tile = blk & 15;
    const int g    = (blk >> 4) & 31;
    const int b    = blk >> 9;
    const int tid  = threadIdx.x;

    const _Float16* vbase = vh + ((size_t)(b * NH_ + g)) * 64 * N_;

    // kick off async staging of V chunk 0 while we stage kp/table normally
    stage_v(vbase, &sV[0][0], tid);

    { // cooperative staging of kp + bias table
        const float* kpb = kp_n + ((size_t)b * N_) * C_ + g * 4;
        for (int mm = tid; mm < N_; mm += 128)
            *(v4f*)&sKP[mm * 4] = *(const v4f*)(kpb + (size_t)mm * C_);
        const float* tb = rel_table + (size_t)g * TBL_;
        for (int ii = tid; ii < TBL_; ii += 128) sTab[ii] = tb[ii];
    }
    __syncthreads();

    const int lane = tid & 31;
    const int wq   = tid >> 5;        // wave id: which 16-query sub-tile
    const int ql   = lane & 15;       // M (query) / N (dp) index within tile
    const int hi   = lane >> 4;       // upper lane half selector

    const int qg = tile * 64 + wq * 16 + ql;     // this lane's query token
    const v4f qv = *(const v4f*)(qp_n + ((size_t)(b * N_ + qg)) * C_ + g * 4);

    v8f acc[4];
#pragma unroll
    for (int dd = 0; dd < 4; ++dd)
#pragma unroll
        for (int e = 0; e < 8; ++e) acc[dd][e] = 0.0f;

    float mrow = -__builtin_inff();
    float lrow = 0.0f;

    const int* ri = rel_index + (size_t)qg * N_;

    for (int mc = 0; mc < N_; mc += 32) {
        const int buf = (mc >> 5) & 1;
        // this lane's 16 keys, in exact A-fragment element order:
        // e<8 : K = mc + 8*hi + e        (run0)
        // e>=8: K = mc + 16 + 8*hi + e-8 (run1)
        const int k0 = mc + hi * 8;
        const int k1 = mc + 16 + hi * 8;

        v4i i0a = *(const v4i*)(ri + k0);
        v4i i0b = *(const v4i*)(ri + k0 + 4);
        v4i i1a = *(const v4i*)(ri + k1);
        v4i i1b = *(const v4i*)(ri + k1 + 4);

        float s[16];
#pragma unroll
        for (int j = 0; j < 8; ++j) {
            const float* kp4 = &sKP[(k0 + j) * 4];
            const int bi = (j < 4) ? i0a[j] : i0b[j - 4];
            s[j] = qv[0]*kp4[0] + qv[1]*kp4[1] + qv[2]*kp4[2] + qv[3]*kp4[3] + sTab[bi];
        }
#pragma unroll
        for (int j = 0; j < 8; ++j) {
            const float* kp4 = &sKP[(k1 + j) * 4];
            const int bi = (j < 4) ? i1a[j] : i1b[j - 4];
            s[8 + j] = qv[0]*kp4[0] + qv[1]*kp4[1] + qv[2]*kp4[2] + qv[3]*kp4[3] + sTab[bi];
        }

        if (mc + 32 < N_) __builtin_prefetch(ri + mc + 32, 0, 1);

        // online softmax: lanes q and q+16 jointly cover all 32 keys of query q
        float mloc = s[0];
#pragma unroll
        for (int j = 1; j < 16; ++j) mloc = fmaxf(mloc, s[j]);
        mloc = fmaxf(mloc, __shfl_xor(mloc, 16, 32));
        const float mnew = fmaxf(mrow, mloc);
        const float sc   = __expf(mrow - mnew);

        float p[16], ls = 0.0f;
#pragma unroll
        for (int j = 0; j < 16; ++j) { p[j] = __expf(s[j] - mnew); ls += p[j]; }
        ls  += __shfl_xor(ls, 16, 32);
        lrow = lrow * sc + ls;
        mrow = mnew;

        // rescale accumulators: row r+8*hi belongs to query (r+8*hi), whose
        // stats live in lane (r+8*hi) -> 8 broadcasts
#pragma unroll
        for (int r = 0; r < 8; ++r) {
            const float srow = __shfl(sc, r + hi * 8, 32);
            acc[0][r] *= srow; acc[1][r] *= srow;
            acc[2][r] *= srow; acc[3][r] *= srow;
        }

        // pack P into A fragment (element e == p[e] by construction)
        union { v16h v; pk2h h[8]; } A;
#pragma unroll
        for (int j2 = 0; j2 < 8; ++j2)
            A.h[j2] = __builtin_amdgcn_cvt_pkrtz(p[2 * j2], p[2 * j2 + 1]);

        // current V chunk is ready once our asyncs landed + all waves arrive
        asm volatile("s_wait_asynccnt 0x0" ::: "memory");
        __syncthreads();
        // overlap: stage next chunk into the other buffer
        if (mc + 32 < N_) stage_v(vbase + (mc + 32), &sV[buf ^ 1][0], tid);

        // 4 B-tiles of V (dp = dd*16 + ql) straight from LDS
#pragma unroll
        for (int dd = 0; dd < 4; ++dd) {
            const _Float16* vrow = &sV[buf][(dd * 16 + ql) * 32 + hi * 8];
            union { v16h v; v8h h[2]; } Bf;
            Bf.h[0] = *(const v8h*)vrow;
            Bf.h[1] = *(const v8h*)(vrow + 16);
            acc[dd] = __builtin_amdgcn_wmma_f32_16x16x32_f16(
                false, A.v, false, Bf.v, (short)0, acc[dd], false, false);
        }
    }

    // epilogue: divide by row sums and scatter back to (B,C,T,H,W)
    const float lrec = 1.0f / lrow;
#pragma unroll
    for (int r = 0; r < 8; ++r) {
        const float li   = __shfl(lrec, r + hi * 8, 32);
        const int  qrow  = tile * 64 + wq * 16 + r + hi * 8;
        const int  tt    = qrow >> 8;
        const int  yh    = (qrow >> 4) & 15;
        const int  xw    = qrow & 15;
        const int  sh    = ql >> 2;
        const int  sw    = ql & 3;
#pragma unroll
        for (int dd = 0; dd < 4; ++dd) {
            const int c = g * 4 + dd;
            size_t oi = ((size_t)((b * C_ + c) * T_ + tt)) * 4096
                      + (yh * 4 + sh) * 64 + (xw * 4 + sw);
            out[oi] = acc[dd][r] * li;
        }
    }
}

// ---------------------------------------------------------------------------
extern "C" void kernel_launch(void* const* d_in, const int* in_sizes, int n_in,
                              void* d_out, int out_size, void* d_ws, size_t ws_size,
                              hipStream_t stream)
{
    const float* x    = (const float*)d_in[0];
    const float* Wq   = (const float*)d_in[1];
    const float* Wk   = (const float*)d_in[2];
    const float* Wv   = (const float*)d_in[3];
    const float* gq_g = (const float*)d_in[4];
    const float* gq_b = (const float*)d_in[5];
    const float* gk_g = (const float*)d_in[6];
    const float* gk_b = (const float*)d_in[7];
    const float* gv_g = (const float*)d_in[8];
    const float* gv_b = (const float*)d_in[9];
    const float* rtab = (const float*)d_in[10];
    const int*   ridx = (const int*)d_in[11];
    float* out = (float*)d_out;

    char* ws = (char*)d_ws;
    size_t off = 0;
    auto take = [&](size_t bytes) -> void* {
        void* p = ws + off;
        off += (bytes + 255) & ~(size_t)255;
        return p;
    };
    float*    qp_raw = (float*)take((size_t)B_ * N_ * C_ * 4);
    float*    kp_raw = (float*)take((size_t)B_ * N_ * C_ * 4);
    float*    qp_n   = (float*)take((size_t)B_ * N_ * C_ * 4);
    float*    kp_n   = (float*)take((size_t)B_ * N_ * C_ * 4);
    _Float16* vh     = (_Float16*)take((size_t)B_ * NH_ * 64 * N_ * 2);
    float*    sraw   = (float*)take(3 * B_ * NH_ * 2 * 4);
    float*    sfin   = (float*)take(3 * B_ * NH_ * 2 * 4);
    (void)ws_size; (void)in_sizes; (void)n_in; (void)out_size;

    k_zero <<<1, 3 * B_ * NH_ * 2, 0, stream>>>(sraw, 3 * B_ * NH_ * 2);
    k_proj <<<B_ * N_, 512, 0, stream>>>(x, Wq, Wk, Wv, qp_raw, kp_raw, vh, sraw);
    k_stats<<<1, 256, 0, stream>>>(sraw, sfin);
    k_norm <<<(B_ * N_ * C_ + 255) / 256, 256, 0, stream>>>(
        qp_raw, kp_raw, gq_g, gq_b, gk_g, gk_b, sfin, qp_n, kp_n);
    k_vnorm<<<(B_ * NH_ * 64 * (N_ / 8) + 255) / 256, 256, 0, stream>>>(
        vh, sfin, gv_g, gv_b);
    k_attn <<<B_ * NH_ * 16, 128, 0, stream>>>(
        qp_n, kp_n, vh, rtab, ridx, out);
}
